// GraphBlock_78451872629215
// MI455X (gfx1250) — compile-verified, matrix operand order checked by reading, not run on hardware
//
#include <hip/hip_runtime.h>
#include <hip/hip_bf16.h>
#include <math.h>

// ---------------------------------------------------------------------------
// CDNA5 (gfx1250) implementation. wave32. WMMA f32<-f16 16x16x32 everywhere
// a GEMM appears. All matrix tiles follow the ISA 7.12.2 VGPR layouts.
// gates_ih is stored time-major ([t][bc][1024]) so the serial LSTM streams
// one contiguous 128KB block per step (plus global_prefetch for step t+1).
// ---------------------------------------------------------------------------

typedef __attribute__((ext_vector_type(16))) _Float16 v16h;
typedef __attribute__((ext_vector_type(8)))  float    v8f;

#define NNODES 100000
#define NEDGES 1600000
#define DEMB   128
#define DMLP   256
#define HDIM   256
#define BC     32      // B*C sequences
#define TLEN   128
#define NPOS   (BC*TLEN)   // 4096 gathered rows
#define G4     (4*HDIM)    // 1024 gate width

// Load one 16x32 f16 operand tile (A layout; B uses the same pattern with
// rowptr = W + (nbase + lane&15)*ldk since B = W^T).
// Lanes 0-15: VGPR v(0..3) -> K = k0 + 2v,2v+1 ; v(4..7) -> K = k0+16+2(v-4)..
// Lanes 16-31: same with +8 on K within each 16-wide half.
__device__ __forceinline__ v16h load_k32(const _Float16* rowptr, int k0, int lane) {
  const int kb = (lane & 16) ? 8 : 0;
  v16h r;
#pragma unroll
  for (int v = 0; v < 4; ++v) {
    int k = k0 + kb + 2 * v;
    r[2 * v]     = rowptr[k];
    r[2 * v + 1] = rowptr[k + 1];
  }
#pragma unroll
  for (int v = 0; v < 4; ++v) {
    int k = k0 + 16 + kb + 2 * v;
    r[8 + 2 * v]     = rowptr[k];
    r[8 + 2 * v + 1] = rowptr[k + 1];
  }
  return r;
}

__device__ __forceinline__ float sigf(float x) { return 1.0f / (1.0f + __expf(-x)); }

// ---------------------------------------------------------------------------
// 0) zero f32 region (agg + cnt)
// ---------------------------------------------------------------------------
__global__ void zero_f32(float* __restrict__ p, int n) {
  for (int i = blockIdx.x * blockDim.x + threadIdx.x; i < n; i += gridDim.x * blockDim.x)
    p[i] = 0.0f;
}

// ---------------------------------------------------------------------------
// 1) f32 -> f16 weight conversion
// ---------------------------------------------------------------------------
__global__ void f32_to_f16(const float* __restrict__ in, _Float16* __restrict__ out, int n) {
  for (int i = blockIdx.x * blockDim.x + threadIdx.x; i < n; i += gridDim.x * blockDim.x)
    out[i] = (_Float16)in[i];
}

// ---------------------------------------------------------------------------
// 2) edge scatter: agg[dst] += node_emb[src]*val ; cnt[dst] += 1
//    one wave per edge, float4 per lane, atomics land in L2 (agg=51MB < 192MB)
// ---------------------------------------------------------------------------
__global__ void edge_scatter(const int* __restrict__ src, const int* __restrict__ dst,
                             const float* __restrict__ ev, const float* __restrict__ nemb,
                             float* __restrict__ agg, float* __restrict__ cnt, int E) {
  long long gid = (long long)blockIdx.x * blockDim.x + threadIdx.x;
  int e = (int)(gid >> 5);
  int lane = (int)(gid & 31);
  if (e >= E) return;
  int s = src[e], d = dst[e];
  float v = ev[e];
  const float4 x = *(const float4*)(nemb + (size_t)s * DEMB + lane * 4);
  float* ap = agg + (size_t)d * DEMB + lane * 4;
  atomicAdd(ap + 0, x.x * v);
  atomicAdd(ap + 1, x.y * v);
  atomicAdd(ap + 2, x.z * v);
  atomicAdd(ap + 3, x.w * v);
  if (lane == 0) atomicAdd(cnt + d, 1.0f);
}

// ---------------------------------------------------------------------------
// 3) gather + mean-finalize only the 4096 rows the LSTM consumes, emit f16
// ---------------------------------------------------------------------------
__global__ void gather_finalize(const int* __restrict__ ids, const float* __restrict__ agg,
                                const float* __restrict__ cnt, _Float16* __restrict__ hg) {
  int p = blockIdx.x;      // 4096
  int d = threadIdx.x;     // 128
  int n = ids[p];
  float c = fmaxf(cnt[n], 1.0f);
  hg[(size_t)p * DEMB + d] = (_Float16)(agg[(size_t)n * DEMB + d] / c);
}

// ---------------------------------------------------------------------------
// 4) WMMA GEMM 1: xg[4096x256] = hg[4096x128] @ W1^T + b1   (f16 out)
//    one wave per 16x16 tile; K = 128 -> 4 wmma
// ---------------------------------------------------------------------------
__global__ void wmma_gemm_xg(const _Float16* __restrict__ A, const _Float16* __restrict__ W,
                             const float* __restrict__ bias, _Float16* __restrict__ out) {
  int wave = blockIdx.x * (blockDim.x >> 5) + (threadIdx.x >> 5);
  int lane = threadIdx.x & 31;
  int mt = wave >> 4;            // 256 M-tiles
  int nt = wave & 15;            // 16 N-tiles
  const _Float16* arow = A + (size_t)(mt * 16 + (lane & 15)) * DEMB;
  const _Float16* brow = W + (size_t)(nt * 16 + (lane & 15)) * DEMB;
  v8f c = {};
#pragma unroll
  for (int k0 = 0; k0 < DEMB; k0 += 32) {
    v16h a = load_k32(arow, k0, lane);
    v16h b = load_k32(brow, k0, lane);
    c = __builtin_amdgcn_wmma_f32_16x16x32_f16(false, a, false, b, (short)0, c, false, false);
  }
  int n = nt * 16 + (lane & 15);
  float bv = bias[n];
  int mbase = mt * 16 + ((lane < 16) ? 0 : 8);
#pragma unroll
  for (int r = 0; r < 8; ++r)
    out[(size_t)(mbase + r) * DMLP + n] = (_Float16)(c[r] + bv);
}

// ---------------------------------------------------------------------------
// 5) WMMA GEMM 2: gates = xg @ w_ih^T + (b_ih + b_hh)  (f32 out)
//    K = 256 -> 8 wmma per tile. Output stored TIME-MAJOR: row m = bc*T + t
//    is written to gates[(t*BC + bc) * 1024 + n] so each LSTM step reads one
//    contiguous 128KB block.
// ---------------------------------------------------------------------------
__global__ void wmma_gemm_gates(const _Float16* __restrict__ A, const _Float16* __restrict__ W,
                                const float* __restrict__ b_ih, const float* __restrict__ b_hh,
                                float* __restrict__ out) {
  int wave = blockIdx.x * (blockDim.x >> 5) + (threadIdx.x >> 5);
  int lane = threadIdx.x & 31;
  int mt = wave >> 6;            // 256 M-tiles
  int nt = wave & 63;            // 64 N-tiles
  const _Float16* arow = A + (size_t)(mt * 16 + (lane & 15)) * DMLP;
  const _Float16* brow = W + (size_t)(nt * 16 + (lane & 15)) * DMLP;
  v8f c = {};
#pragma unroll
  for (int k0 = 0; k0 < DMLP; k0 += 32) {
    v16h a = load_k32(arow, k0, lane);
    v16h b = load_k32(brow, k0, lane);
    c = __builtin_amdgcn_wmma_f32_16x16x32_f16(false, a, false, b, (short)0, c, false, false);
  }
  int n = nt * 16 + (lane & 15);
  float bv = b_ih[n] + b_hh[n];
  int mbase = mt * 16 + ((lane < 16) ? 0 : 8);
#pragma unroll
  for (int r = 0; r < 8; ++r) {
    int m  = mbase + r;           // = bc*TLEN + t
    int bc = m >> 7;              // m / TLEN
    int t  = m & (TLEN - 1);      // m % TLEN
    out[(size_t)(t * BC + bc) * G4 + n] = c[r] + bv;
  }
}

// ---------------------------------------------------------------------------
// 6) LSTM: single workgroup, 8 waves. h state f16 in LDS (double buffered),
//    c state in registers. Per step, per wave: 4 groups x 4 gate tiles x
//    8 wmma (K=256). w_hh (f16, 512KB) streams from L2 each step; the next
//    step's contiguous gates block is prefetched (global_prefetch_b8) while
//    this step's WMMAs run.
// ---------------------------------------------------------------------------
__global__ void lstm_kernel(const float* __restrict__ gates, const _Float16* __restrict__ Whh,
                            const int* __restrict__ mask, float* __restrict__ hlast) {
  __shared__ _Float16 hbuf[2][BC][HDIM];   // 32 KB
  __shared__ int lens[BC];
  int tid = threadIdx.x;
  int lane = tid & 31;
  int wave = tid >> 5;                     // 0..7

  for (int i = tid; i < 2 * BC * HDIM; i += blockDim.x)
    ((_Float16*)hbuf)[i] = (_Float16)0.0f;
  if (tid < BC) {
    int s = 0;
    for (int t = 0; t < TLEN; ++t) s += (mask[tid * TLEN + t] != 0) ? 1 : 0;
    lens[tid] = s;
  }
  __syncthreads();

  float cst[4][8];
#pragma unroll
  for (int i = 0; i < 4; ++i)
#pragma unroll
    for (int r = 0; r < 8; ++r) cst[i][r] = 0.0f;

  int cur = 0;
  for (int t = 0; t < TLEN; ++t) {
    // prefetch next step's 128KB gates block (256 cachelines; 256 threads)
    if (t + 1 < TLEN) {
      const char* nxt = (const char*)(gates + (size_t)(t + 1) * BC * G4);
      __builtin_prefetch(nxt + (size_t)tid * 512, 0, 1);
    }
#pragma unroll
    for (int i = 0; i < 4; ++i) {
      int g = wave * 4 + i;                // 0..31
      int mt = g & 1;                      // row tile (seqs 0-15 / 16-31)
      int j  = g >> 1;                     // h-column tile 0..15
      const _Float16* arow = &hbuf[cur][mt * 16 + (lane & 15)][0];
      int mbase = mt * 16 + ((lane < 16) ? 0 : 8);
      int hcol  = j * 16 + (lane & 15);

      v8f acc[4];
#pragma unroll
      for (int q = 0; q < 4; ++q) {        // i,f,g,o gate tiles
        int col = q * HDIM + hcol;
        v8f c;
#pragma unroll
        for (int r = 0; r < 8; ++r)
          c[r] = gates[(size_t)(t * BC + mbase + r) * G4 + col];
        acc[q] = c;
      }
#pragma unroll
      for (int k0 = 0; k0 < HDIM; k0 += 32) {
        v16h a = load_k32(arow, k0, lane);
#pragma unroll
        for (int q = 0; q < 4; ++q) {
          const _Float16* brow = Whh + (size_t)(q * HDIM + hcol) * HDIM;
          v16h b = load_k32(brow, k0, lane);
          acc[q] = __builtin_amdgcn_wmma_f32_16x16x32_f16(false, a, false, b,
                                                          (short)0, acc[q], false, false);
        }
      }
#pragma unroll
      for (int r = 0; r < 8; ++r) {
        int bc = mbase + r;
        float iv = sigf(acc[0][r]);
        float fv = sigf(acc[1][r]);
        float gv = tanhf(acc[2][r]);
        float ov = sigf(acc[3][r]);
        float cn = fv * cst[i][r] + iv * gv;
        float hn = ov * tanhf(cn);
        cst[i][r] = cn;
        hbuf[cur ^ 1][bc][hcol] = (_Float16)hn;
        if (t == lens[bc] - 1) hlast[bc * HDIM + hcol] = hn;
      }
    }
    __syncthreads();
    cur ^= 1;
  }
}

// ---------------------------------------------------------------------------
// 7) head: scores -> softmax -> error + argmax
// ---------------------------------------------------------------------------
__global__ void head_kernel(const float* __restrict__ hlast, const float* __restrict__ W2,
                            const float* __restrict__ b2, const float* __restrict__ labels,
                            float* __restrict__ out) {
  __shared__ float sc[BC];
  __shared__ float errs[8];
  int tid = threadIdx.x;
  if (tid < BC) {
    float s = b2[0];
    for (int k = 0; k < HDIM; ++k) s += hlast[tid * HDIM + k] * W2[k];
    sc[tid] = s;
  }
  __syncthreads();
  if (tid < 8) {
    float m = sc[tid * 4];
    for (int c = 1; c < 4; ++c) m = fmaxf(m, sc[tid * 4 + c]);
    float e[4], sum = 0.0f;
    for (int c = 0; c < 4; ++c) { e[c] = __expf(sc[tid * 4 + c] - m); sum += e[c]; }
    float err = 0.0f, best = -1.0f;
    int arg = 0;
    for (int c = 0; c < 4; ++c) {
      float p = e[c] / sum;
      if (p > best) { best = p; arg = c; }
      float l = labels[tid * 4 + c];
      err += fmaxf(p, 0.0f) - p * l + log1pf(__expf(-fabsf(p)));
    }
    errs[tid] = err;
    out[1 + tid] = (float)arg;
  }
  __syncthreads();
  if (tid == 0) {
    float s = 0.0f;
    for (int b = 0; b < 8; ++b) s += errs[b];
    out[0] = s / 32.0f;
  }
}

// ---------------------------------------------------------------------------
extern "C" void kernel_launch(void* const* d_in, const int* in_sizes, int n_in,
                              void* d_out, int out_size, void* d_ws, size_t ws_size,
                              hipStream_t stream) {
  const int*   ids    = (const int*)d_in[0];
  const int*   mask   = (const int*)d_in[1];
  const float* labels = (const float*)d_in[2];
  const int*   src    = (const int*)d_in[3];
  const int*   dst    = (const int*)d_in[4];
  const float* ev     = (const float*)d_in[5];
  const float* nemb   = (const float*)d_in[6];
  const float* W1     = (const float*)d_in[7];
  const float* b1     = (const float*)d_in[8];
  const float* w_ih   = (const float*)d_in[9];
  const float* w_hh   = (const float*)d_in[10];
  const float* b_ih   = (const float*)d_in[11];
  const float* b_hh   = (const float*)d_in[12];
  const float* W2     = (const float*)d_in[13];
  const float* b2     = (const float*)d_in[14];
  float* out = (float*)d_out;

  char* ws = (char*)d_ws;
  size_t off = 0;
  auto take = [&](size_t bytes) -> char* {
    char* p = ws + off;
    off = (off + bytes + 255) & ~(size_t)255;
    return p;
  };
  float*     agg    = (float*)take((size_t)NNODES * DEMB * 4);   // 51.2 MB
  float*     cnt    = (float*)take((size_t)NNODES * 4);
  _Float16*  W1h    = (_Float16*)take((size_t)DMLP * DEMB * 2);
  _Float16*  wihh   = (_Float16*)take((size_t)G4 * DMLP * 2);
  _Float16*  whhh   = (_Float16*)take((size_t)G4 * HDIM * 2);
  _Float16*  hg     = (_Float16*)take((size_t)NPOS * DEMB * 2);
  _Float16*  xg     = (_Float16*)take((size_t)NPOS * DMLP * 2);
  float*     gates  = (float*)take((size_t)NPOS * G4 * 4);       // 16.8 MB, [t][bc][1024]
  float*     hlast  = (float*)take((size_t)BC * HDIM * 4);
  (void)ws_size; (void)in_sizes; (void)n_in; (void)out_size;

  // 0) zero accumulators
  zero_f32<<<2048, 256, 0, stream>>>(agg, NNODES * DEMB);
  zero_f32<<<64, 256, 0, stream>>>(cnt, NNODES);

  // 1) weight precision conversion (f32 -> f16 for WMMA)
  f32_to_f16<<<128, 256, 0, stream>>>(W1, W1h, DMLP * DEMB);
  f32_to_f16<<<512, 256, 0, stream>>>(w_ih, wihh, G4 * DMLP);
  f32_to_f16<<<512, 256, 0, stream>>>(w_hh, whhh, G4 * HDIM);

  // 2) edge-weighted scatter-mean numerators/denominators
  {
    long long threads = (long long)NEDGES * 32;
    int blocks = (int)((threads + 255) / 256);
    edge_scatter<<<blocks, 256, 0, stream>>>(src, dst, ev, nemb, agg, cnt, NEDGES);
  }

  // 3) gather + finalize only the 4096 LSTM-consumed rows
  gather_finalize<<<NPOS, DEMB, 0, stream>>>(ids, agg, cnt, hg);

  // 4) xg = hg @ W1^T + b1        (4096x128 @ 128x256), 4096 tiles, 8 waves/blk
  wmma_gemm_xg<<<512, 256, 0, stream>>>(hg, W1h, b1, xg);

  // 5) gates_ih = xg @ w_ih^T + b (4096x256 @ 256x1024), 16384 tiles, time-major out
  wmma_gemm_gates<<<2048, 256, 0, stream>>>(xg, wihh, b_ih, b_hh, gates);

  // 6) recurrent LSTM (serial over T, WMMA for h @ w_hh^T)
  lstm_kernel<<<1, 256, 0, stream>>>(gates, whhh, mask, hlast);

  // 7) head: scores, softmax, BCE-style error, argmax
  head_kernel<<<1, 64, 0, stream>>>(hlast, W2, b2, labels, out);
}
